// Net_77163382440206
// MI455X (gfx1250) — matrix operand order repaired and checked by reading, not compiled, not run on hardware
//
#include <hip/hip_runtime.h>
#include <math.h>

typedef _Float16 half_t;
typedef _Float16 v16h __attribute__((ext_vector_type(16)));
typedef float    v8f  __attribute__((ext_vector_type(8)));

#define NN   50000
#define EE   600000
#define EPN  650000   // EE + NN
#define FIN  128
#define H1C  512
#define HIDC 64

enum { EPI_NONE = 0, EPI_LEAKY02 = 1, EPI_BIAS = 2, EPI_COMBINE = 3 };

// ---------------- B pre-pack: f32 [K,Nc] -> f16 WMMA fragments ----------------
// Fragment layout (ISA §7.12.2, 32x16 f16 B tile): element e of lane L in tile (kt,nt):
//   k = kt*32 + (e&8)*2 + (L>>4)*8 + (e&7),  n = nt*16 + (L&15);  zero-padded for k >= K.
// Packed offset: (((kt*nTiles + nt)*32 + L)*16 + e)
__global__ void k_pack_b(const float* __restrict__ B, half_t* __restrict__ P,
                         int K, int Nc) {
  int nTiles = Nc >> 4;
  int kTiles = (K + 31) >> 5;
  long total = (long)kTiles * nTiles * 512;  // 32 lanes * 16 elems
  long tid = blockIdx.x * (long)blockDim.x + threadIdx.x;
  if (tid >= total) return;
  int e    = (int)(tid & 15);
  int lane = (int)((tid >> 4) & 31);
  long blk = tid >> 9;                        // kt*nTiles + nt
  int kt = (int)(blk / nTiles), nt = (int)(blk % nTiles);
  int k = kt * 32 + (e & 8) * 2 + (lane >> 4) * 8 + (e & 7);
  int n = nt * 16 + (lane & 15);
  P[tid] = (k < K) ? (half_t)B[(long)k * Nc + n] : (half_t)0.0f;
}

// ---------------- A fragment loader (16x32 f16, ISA §7.12.2) ----------------
template <int K>
__device__ __forceinline__ v16h load_a_frag(const float* __restrict__ A,
                                            int row0, int k0, int lane) {
  int m  = row0 + (lane & 15);
  int hf = lane >> 4;
  const float* Ar = A + (long)m * K + k0 + hf * 8;
  float4 c0 = *(const float4*)(Ar + 0);
  float4 c1 = *(const float4*)(Ar + 4);
  float4 c2, c3;
  if constexpr (K >= 32) {
    c2 = *(const float4*)(Ar + 16);
    c3 = *(const float4*)(Ar + 20);
  } else {
    c2 = make_float4(0.f, 0.f, 0.f, 0.f);
    c3 = make_float4(0.f, 0.f, 0.f, 0.f);
  }
  v16h a;
  a[0]  = (_Float16)c0.x; a[1]  = (_Float16)c0.y; a[2]  = (_Float16)c0.z; a[3]  = (_Float16)c0.w;
  a[4]  = (_Float16)c1.x; a[5]  = (_Float16)c1.y; a[6]  = (_Float16)c1.z; a[7]  = (_Float16)c1.w;
  a[8]  = (_Float16)c2.x; a[9]  = (_Float16)c2.y; a[10] = (_Float16)c2.z; a[11] = (_Float16)c2.w;
  a[12] = (_Float16)c3.x; a[13] = (_Float16)c3.y; a[14] = (_Float16)c3.z; a[15] = (_Float16)c3.w;
  return a;
}

// ---------------- Generic WMMA GEMM (pre-packed B, software-pipelined) ----------------
// C[M,Nc] = epilogue(A[M,K] @ B[K,Nc]);  M%16==0, Nc%(16*NT)==0, K compile-time.
template <int EPI, int NT, int K>
__global__ __launch_bounds__(128) void wmma_gemm_kernel(
    const float* __restrict__ A, const half_t* __restrict__ Bp, float* __restrict__ C,
    int M, int Nc,
    const float* __restrict__ bias,   // col bias (EPI_BIAS) or bias1 (EPI_COMBINE)
    const float* __restrict__ pbias,  // p_b2     (EPI_COMBINE)
    const float* __restrict__ aux,    // agg      (EPI_COMBINE, read-modify same as C)
    const float* __restrict__ alphaP) {
  int lane = threadIdx.x & 31;
  int wave = (int)((blockIdx.x * (long)blockDim.x + threadIdx.x) >> 5);
  int mTiles  = M >> 4;
  int nTiles  = Nc >> 4;
  int nSupers = Nc / (16 * NT);
  if (wave >= mTiles * nSupers) return;  // uniform per wave -> EXEC all ones below
  int mTile = wave / nSupers, nSuper = wave % nSupers;
  int row0 = mTile << 4;
  int nt0  = nSuper * NT;

  v8f acc[NT];
#pragma unroll
  for (int t = 0; t < NT; ++t)
#pragma unroll
    for (int r = 0; r < 8; ++r) acc[t][r] = 0.0f;

  long bStep = (long)nTiles * 512;            // f16 elems per K-step
  const half_t* bptr = Bp + (long)nt0 * 512 + (long)lane * 16;

  // prologue: fragments for k0 = 0
  v16h a_cur = load_a_frag<K>(A, row0, 0, lane);
  v16h b_cur[NT];
#pragma unroll
  for (int t = 0; t < NT; ++t) b_cur[t] = *(const v16h*)(bptr + (long)t * 512);

  // pipelined steady state: prefetch k0+32 while multiplying k0
#pragma unroll 1
  for (int k0 = 0; k0 + 32 < K; k0 += 32) {
    v16h a_nxt = load_a_frag<K>(A, row0, k0 + 32, lane);
    bptr += bStep;
    v16h b_nxt[NT];
#pragma unroll
    for (int t = 0; t < NT; ++t) b_nxt[t] = *(const v16h*)(bptr + (long)t * 512);
#pragma unroll
    for (int t = 0; t < NT; ++t)
      acc[t] = __builtin_amdgcn_wmma_f32_16x16x32_f16(
          false, a_cur, false, b_cur[t], (short)0, acc[t], false, false);
    a_cur = a_nxt;
#pragma unroll
    for (int t = 0; t < NT; ++t) b_cur[t] = b_nxt[t];
  }
  // final K-step
#pragma unroll
  for (int t = 0; t < NT; ++t)
    acc[t] = __builtin_amdgcn_wmma_f32_16x16x32_f16(
        false, a_cur, false, b_cur[t], (short)0, acc[t], false, false);

  float alpha = 0.0f;
  if (EPI == EPI_COMBINE) alpha = alphaP[0];
  int nbase = lane & 15, hf = lane >> 4;
#pragma unroll
  for (int t = 0; t < NT; ++t) {
    int n = (nt0 + t) * 16 + nbase;
#pragma unroll
    for (int r = 0; r < 8; ++r) {
      int  m   = row0 + (hf << 3) + r;
      long idx = (long)m * Nc + n;
      float v  = acc[t][r];
      if (EPI == EPI_LEAKY02) {
        v = v > 0.0f ? v : 0.2f * v;
      } else if (EPI == EPI_BIAS) {
        v += bias[n];
      } else if (EPI == EPI_COMBINE) {
        float p = v + pbias[n];
        p = p > 0.0f ? p : 0.01f * p;                 // leaky 0.01
        float o = aux[idx] + bias[n] + alpha * p;     // agg + bias1 + alpha*p
        v = o > 0.0f ? o : 0.0f;                      // relu between layers
      }
      C[idx] = v;
    }
  }
}

// ---------------- Graph kernels ----------------
__device__ __forceinline__ int src_of(const int* __restrict__ ei, int e) {
  return e < EE ? ei[e] : e - EE;
}
__device__ __forceinline__ int dst_of(const int* __restrict__ ei, int e) {
  return e < EE ? ei[EE + e] : e - EE;
}

__device__ __forceinline__ void atomicMaxF32(float* addr, float val) {
  unsigned int* ua  = (unsigned int*)addr;
  unsigned int  cur = *ua;
  while (__uint_as_float(cur) < val) {
    unsigned int prev = atomicCAS(ua, cur, __float_as_uint(val));
    if (prev == cur) break;
    cur = prev;
  }
}

__global__ void k_fill(float* __restrict__ p, long n, float v) {
  long i = blockIdx.x * (long)blockDim.x + threadIdx.x;
  if (i < n) p[i] = v;
}

__global__ void k_segmax(const float* __restrict__ w, const int* __restrict__ ei,
                         float* __restrict__ mx, int C) {
  long tid = blockIdx.x * (long)blockDim.x + threadIdx.x;
  long total = (long)EPN * C;
  if (tid >= total) return;
  int c = (int)(tid % C), e = (int)(tid / C);
  int s = src_of(ei, e);
  atomicMaxF32(&mx[(long)s * C + c], w[tid]);
}

__global__ void k_expsum(float* __restrict__ w, const int* __restrict__ ei,
                         const float* __restrict__ mx, float* __restrict__ sum, int C) {
  long tid = blockIdx.x * (long)blockDim.x + threadIdx.x;
  long total = (long)EPN * C;
  if (tid >= total) return;
  int c = (int)(tid % C), e = (int)(tid / C);
  int s = src_of(ei, e);
  float ev = expf(w[tid] - mx[(long)s * C + c]);
  w[tid] = ev;
  atomicAdd(&sum[(long)s * C + c], ev);
}

__global__ void k_norm(float* __restrict__ w, const int* __restrict__ ei,
                       const float* __restrict__ sum, int C) {
  long tid = blockIdx.x * (long)blockDim.x + threadIdx.x;
  long total = (long)EPN * C;
  if (tid >= total) return;
  int c = (int)(tid % C), e = (int)(tid / C);
  int s = src_of(ei, e);
  w[tid] = w[tid] / (sum[(long)s * C + c] + 1e-16f);
}

// msg = a[e, c%Ca] * xh[src, c]; agg[dst, c] += msg    (float4 per thread)
__global__ void k_scatter(const float* __restrict__ aw, const float* __restrict__ xh,
                          const int* __restrict__ ei, float* __restrict__ agg,
                          int Ctot, int Ca) {
  long tid = blockIdx.x * (long)blockDim.x + threadIdx.x;
  int perRow = Ctot >> 2;
  long total = (long)EPN * perRow;
  if (tid >= total) return;
  int e = (int)(tid / perRow);
  int q = (int)(tid % perRow) << 2;
  int s = src_of(ei, e), d = dst_of(ei, e);
  int ca = q % Ca;
  float4 xv = *(const float4*)(xh + (long)s * Ctot + q);
  float4 av = *(const float4*)(aw + (long)e * Ca + ca);
  float* dst = agg + (long)d * Ctot + q;
  atomicAdd(dst + 0, av.x * xv.x);
  atomicAdd(dst + 1, av.y * xv.y);
  atomicAdd(dst + 2, av.z * xv.z);
  atomicAdd(dst + 3, av.w * xv.w);
}

// Layer-2 tail: per-node 16x16 MLP on e_poinc + combine + log_softmax
__global__ void k_final(const float* __restrict__ agg2, const float* __restrict__ ep,
                        const float* __restrict__ w1, const float* __restrict__ w2,
                        const float* __restrict__ pb2, const float* __restrict__ b2,
                        const float* __restrict__ alphaP, float* __restrict__ out) {
  __shared__ float w1s[256], w2s[256], pbs[16], b2s[16];
  int t = threadIdx.x;
  if (t < 256) { w1s[t] = w1[t]; w2s[t] = w2[t]; }
  if (t < 16)  { pbs[t] = pb2[t]; b2s[t] = b2[t]; }
  __syncthreads();
  int n = blockIdx.x * blockDim.x + t;
  if (n >= NN) return;
  float alpha = alphaP[0];
  float e[16], mid[16], l[16];
#pragma unroll
  for (int j = 0; j < 16; ++j) e[j] = ep[(long)n * 16 + j];
#pragma unroll
  for (int c = 0; c < 16; ++c) {
    float s = 0.0f;
#pragma unroll
    for (int j = 0; j < 16; ++j) s += e[j] * w1s[j * 16 + c];
    mid[c] = s > 0.0f ? s : 0.2f * s;
  }
#pragma unroll
  for (int c = 0; c < 16; ++c) {
    float v = pbs[c];
#pragma unroll
    for (int j = 0; j < 16; ++j) v += mid[j] * w2s[j * 16 + c];
    float p = v > 0.0f ? v : 0.01f * v;
    l[c] = agg2[(long)n * 16 + c] + b2s[c] + alpha * p;
  }
  float m = l[0];
#pragma unroll
  for (int c = 1; c < 16; ++c) m = fmaxf(m, l[c]);
  float s = 0.0f;
#pragma unroll
  for (int c = 0; c < 16; ++c) s += expf(l[c] - m);
  float lse = m + logf(s);
#pragma unroll
  for (int c = 0; c < 16; ++c) out[(long)n * 16 + c] = l[c] - lse;
}

// ---------------- host ----------------
static inline int fill_blocks(long n) { return (int)((n + 255) / 256); }
static inline int gemm_blocks(long M, int Nc, int NT) {
  long waves = (M / 16) * ((long)Nc / (16 * NT));
  return (int)((waves * 32 + 127) / 128);
}
static inline long packed_elems(int K, int Nc) {
  return (long)((K + 31) / 32) * 32 * Nc;  // Kpad * Nc f16 elements
}

extern "C" void kernel_launch(void* const* d_in, const int* in_sizes, int n_in,
                              void* d_out, int out_size, void* d_ws, size_t ws_size,
                              hipStream_t stream) {
  const float* x       = (const float*)d_in[0];
  const int*   ei      = (const int*)  d_in[1];
  const float* alphaP  = (const float*)d_in[2];
  const float* k_ricci = (const float*)d_in[3];
  const float* e_poinc = (const float*)d_in[4];
  const float* W_src1  = (const float*)d_in[5];
  const float* h1_w1   = (const float*)d_in[6];
  const float* h1_w2   = (const float*)d_in[7];
  const float* h1_b2   = (const float*)d_in[8];
  const float* p1_w1   = (const float*)d_in[9];
  const float* p1_w2   = (const float*)d_in[10];
  const float* p1_b2   = (const float*)d_in[11];
  const float* bias1   = (const float*)d_in[12];
  const float* W_src2  = (const float*)d_in[13];
  const float* h2_w1   = (const float*)d_in[14];
  const float* h2_w2   = (const float*)d_in[15];
  const float* h2_b2   = (const float*)d_in[16];
  const float* p2_w1   = (const float*)d_in[17];
  const float* p2_w2   = (const float*)d_in[18];
  const float* p2_b2   = (const float*)d_in[19];
  const float* bias2   = (const float*)d_in[20];
  float* out = (float*)d_out;

  // workspace layout (floats)
  float* t0 = (float*)d_ws;                  // N*512 : xh1 -> pmid -> agg2/xh2
  float* t1 = t0 + (long)NN * H1C;           // EP*64 : mid1 -> mid2+ew2
  float* t2 = t1 + (long)EPN * HIDC;         // EP*64 : edge_w1 / a1
  float* sm = t2 + (long)EPN * HIDC;         // N*64  : segmax
  float* ss = sm + (long)NN * HIDC;          // N*64  : segsum
  float* t5 = ss + (long)NN * HIDC;          // N*512 : agg1 -> h (layer1 output)

  // packed f16 weight fragments (after t5; all offsets 32B-aligned)
  half_t* pWs1  = (half_t*)(t5 + (long)NN * H1C);
  half_t* pH1w1 = pWs1  + packed_elems(FIN, H1C);   // 128x512
  half_t* pH1w2 = pH1w1 + packed_elems(16, HIDC);   // pad32 x 64
  half_t* pP1w1 = pH1w2 + packed_elems(HIDC, HIDC); // 64x64
  half_t* pP1w2 = pP1w1 + packed_elems(16, H1C);    // pad32 x 512
  half_t* pWs2  = pP1w2 + packed_elems(H1C, H1C);   // 512x512
  half_t* pH2w1 = pWs2  + packed_elems(H1C, 16);    // 512x16
  half_t* pH2w2 = pH2w1 + packed_elems(16, 16);     // pad32 x 16

  // ---- pack all B matrices into WMMA fragment layout ----
  k_pack_b<<<fill_blocks(packed_elems(FIN, H1C)), 256, 0, stream>>>(W_src1, pWs1, FIN, H1C);
  k_pack_b<<<fill_blocks(packed_elems(16, HIDC)), 256, 0, stream>>>(h1_w1, pH1w1, 16, HIDC);
  k_pack_b<<<fill_blocks(packed_elems(HIDC, HIDC)), 256, 0, stream>>>(h1_w2, pH1w2, HIDC, HIDC);
  k_pack_b<<<fill_blocks(packed_elems(16, H1C)), 256, 0, stream>>>(p1_w1, pP1w1, 16, H1C);
  k_pack_b<<<fill_blocks(packed_elems(H1C, H1C)), 256, 0, stream>>>(p1_w2, pP1w2, H1C, H1C);
  k_pack_b<<<fill_blocks(packed_elems(H1C, 16)), 256, 0, stream>>>(W_src2, pWs2, H1C, 16);
  k_pack_b<<<fill_blocks(packed_elems(16, 16)), 256, 0, stream>>>(h2_w1, pH2w1, 16, 16);
  k_pack_b<<<fill_blocks(packed_elems(16, 16)), 256, 0, stream>>>(h2_w2, pH2w2, 16, 16);

  // ================= Layer 1 =================
  k_fill<<<fill_blocks((long)NN * H1C), 256, 0, stream>>>(t5, (long)NN * H1C, 0.0f);
  k_fill<<<fill_blocks((long)NN * HIDC), 256, 0, stream>>>(sm, (long)NN * HIDC, -INFINITY);
  k_fill<<<fill_blocks((long)NN * HIDC), 256, 0, stream>>>(ss, (long)NN * HIDC, 0.0f);

  // xh1 = x @ W_src1                       [50000,128]x[128,512]
  wmma_gemm_kernel<EPI_NONE, 4, FIN><<<gemm_blocks(NN, 512, 4), 128, 0, stream>>>(
      x, pWs1, t0, NN, 512, nullptr, nullptr, nullptr, nullptr);
  // mid1 = leaky(k_ricci @ h1_w1, 0.2)     [650000,16]x[16,64]  (K padded 16->32)
  wmma_gemm_kernel<EPI_LEAKY02, 4, 16><<<gemm_blocks(EPN, 64, 4), 128, 0, stream>>>(
      k_ricci, pH1w1, t1, EPN, 64, nullptr, nullptr, nullptr, nullptr);
  // edge_w1 = mid1 @ h1_w2 + h1_b2         [650000,64]x[64,64]
  wmma_gemm_kernel<EPI_BIAS, 4, HIDC><<<gemm_blocks(EPN, 64, 4), 128, 0, stream>>>(
      t1, pH1w2, t2, EPN, 64, h1_b2, nullptr, nullptr, nullptr);

  long ec1 = (long)EPN * HIDC;
  k_segmax<<<fill_blocks(ec1), 256, 0, stream>>>(t2, ei, sm, HIDC);
  k_expsum<<<fill_blocks(ec1), 256, 0, stream>>>(t2, ei, sm, ss, HIDC);
  k_norm<<<fill_blocks(ec1), 256, 0, stream>>>(t2, ei, ss, HIDC);
  // agg1[dst] += a1 * xh1[src]
  k_scatter<<<fill_blocks((long)EPN * (H1C / 4)), 256, 0, stream>>>(
      t2, t0, ei, t5, H1C, HIDC);

  // pmid = leaky(e_poinc @ p1_w1, 0.2)     [50000,16]x[16,512]
  wmma_gemm_kernel<EPI_LEAKY02, 4, 16><<<gemm_blocks(NN, 512, 4), 128, 0, stream>>>(
      e_poinc, pP1w1, t0, NN, 512, nullptr, nullptr, nullptr, nullptr);
  // h = relu(agg1 + bias1 + alpha*leaky(pmid @ p1_w2 + p1_b2, 0.01))  (in place on t5)
  wmma_gemm_kernel<EPI_COMBINE, 4, H1C><<<gemm_blocks(NN, 512, 4), 128, 0, stream>>>(
      t0, pP1w2, t5, NN, 512, bias1, p1_b2, t5, alphaP);

  // ================= Layer 2 =================
  float* agg2 = t0;
  float* xh2  = t0 + (long)NN * 16;
  float* mid2 = t1;
  float* ew2  = t1 + (long)EPN * 16;
  float* sm2  = sm;
  float* ss2  = sm + (long)NN * 16;

  k_fill<<<fill_blocks((long)NN * 16), 256, 0, stream>>>(agg2, (long)NN * 16, 0.0f);
  k_fill<<<fill_blocks((long)NN * 16), 256, 0, stream>>>(sm2, (long)NN * 16, -INFINITY);
  k_fill<<<fill_blocks((long)NN * 16), 256, 0, stream>>>(ss2, (long)NN * 16, 0.0f);

  // xh2 = h @ W_src2                       [50000,512]x[512,16]
  wmma_gemm_kernel<EPI_NONE, 1, H1C><<<gemm_blocks(NN, 16, 1), 128, 0, stream>>>(
      t5, pWs2, xh2, NN, 16, nullptr, nullptr, nullptr, nullptr);
  // mid2 = leaky(k_ricci @ h2_w1, 0.2)     [650000,16]x[16,16]
  wmma_gemm_kernel<EPI_LEAKY02, 1, 16><<<gemm_blocks(EPN, 16, 1), 128, 0, stream>>>(
      k_ricci, pH2w1, mid2, EPN, 16, nullptr, nullptr, nullptr, nullptr);
  // ew2 = mid2 @ h2_w2 + h2_b2             [650000,16]x[16,16]
  wmma_gemm_kernel<EPI_BIAS, 1, 16><<<gemm_blocks(EPN, 16, 1), 128, 0, stream>>>(
      mid2, pH2w2, ew2, EPN, 16, h2_b2, nullptr, nullptr, nullptr);

  long ec2 = (long)EPN * 16;
  k_segmax<<<fill_blocks(ec2), 256, 0, stream>>>(ew2, ei, sm2, 16);
  k_expsum<<<fill_blocks(ec2), 256, 0, stream>>>(ew2, ei, sm2, ss2, 16);
  k_norm<<<fill_blocks(ec2), 256, 0, stream>>>(ew2, ei, ss2, 16);
  k_scatter<<<fill_blocks((long)EPN * 4), 256, 0, stream>>>(
      ew2, xh2, ei, agg2, 16, 16);

  // log_softmax(agg2 + bias2 + alpha * ham(e_poinc))
  k_final<<<(NN + 255) / 256, 256, 0, stream>>>(
      agg2, e_poinc, p2_w1, p2_w2, p2_b2, bias2, alphaP, out);

  (void)in_sizes; (void)n_in; (void)out_size; (void)ws_size;
}
